// DeepseekV2AttentionVanilla_20727512170704
// MI455X (gfx1250) — matrix operand order repaired and checked by reading, not compiled, not run on hardware
//
#include <hip/hip_runtime.h>
#include <hip/hip_bf16.h>

// ---------------------------------------------------------------------------
// DeepseekV2 MLA decode, absorbed form, for gfx1250 (MI455X, wave32, WMMA).
// b=2, kv_len=2048, H=5120, NH=128, QLR=1536, KVLR=512, DN=128, DR=64, DV=128.
// Bandwidth-bound on fp32 weights (~585 MB -> ~25us @ 23.3 TB/s). The two
// dense GEMMs (scores: 128x2048 K=576, ctx: 128x512 K=2048) use
// v_wmma_f32_16x16x32_bf16 with all fragments loaded as 128-bit chunks
// (ctx uses a kv-major transposed latent cache so B is K-contiguous).
// ---------------------------------------------------------------------------

typedef __attribute__((ext_vector_type(16))) __bf16 v16bf;
typedef __attribute__((ext_vector_type(8)))  __bf16 v8bf;
typedef __attribute__((ext_vector_type(8)))  float  v8f;
typedef __bf16 bf16_t;

#define NH    128
#define HDIM  5120
#define QLR   1536
#define KVLR  512
#define DN    128
#define DR    64
#define DV    128
#define DQ    192
#define KVLEN 2048
#define BATCH 2
#define KCAT  (KVLR + DR)   // 576 = absorbed-q / latent-k feature dim

__device__ __forceinline__ bf16_t f2bf(float f) {
  unsigned u = __float_as_uint(f);
  u += 0x7fffu + ((u >> 16) & 1u);          // round-to-nearest-even
  unsigned short h = (unsigned short)(u >> 16);
  return __builtin_bit_cast(bf16_t, h);
}

__device__ __forceinline__ float wave_sum(float v) {
#pragma unroll
  for (int m = 16; m > 0; m >>= 1) v += __shfl_xor(v, m, 32);
  return v;
}

// ---------------------------------------------------------------------------
// Generic wave-per-row GEMV: out[b*rowsPerB + r] = w[r,:] . x[b,:]
// K must be a multiple of 128. 256 threads = 8 wave32 rows per block.
// ---------------------------------------------------------------------------
__global__ __launch_bounds__(256) void gemv_rows(
    const float* __restrict__ x, const float* __restrict__ w,
    float* __restrict__ out, int rowsPerB, int K) {
  const int row  = blockIdx.x * 8 + (threadIdx.x >> 5);
  const int lane = threadIdx.x & 31;
  const int b    = row / rowsPerB;
  const int r    = row - b * rowsPerB;
  const float* wrow = w + (size_t)r * K;
  const float* xrow = x + (size_t)b * K;
  float s = 0.f;
  for (int k = lane * 4; k < K; k += 128) {
    if (k + 4096 < K) __builtin_prefetch(wrow + k + 4096, 0, 1); // global_prefetch_b8
    float4 wv = *(const float4*)(wrow + k);
    float4 xv = *(const float4*)(xrow + k);
    s += wv.x * xv.x + wv.y * xv.y + wv.z * xv.z + wv.w * xv.w;
  }
  s = wave_sum(s);
  if (lane == 0) out[row] = s;
}

// ---------------------------------------------------------------------------
// RMSNorm over QLR per batch row.
// ---------------------------------------------------------------------------
__global__ __launch_bounds__(256) void rmsnorm_k(
    const float* __restrict__ qa, const float* __restrict__ w,
    float* __restrict__ out) {
  const int b = blockIdx.x, tid = threadIdx.x;
  const int lane = tid & 31, wid = tid >> 5;
  const float* x = qa + (size_t)b * QLR;
  float ss = 0.f;
  for (int i = tid; i < QLR; i += 256) { float v = x[i]; ss += v * v; }
  __shared__ float red[8]; __shared__ float bval;
  ss = wave_sum(ss);
  if (lane == 0) red[wid] = ss;
  __syncthreads();
  if (tid == 0) { float t = 0.f; for (int j = 0; j < 8; ++j) t += red[j]; bval = t; }
  __syncthreads();
  const float rs = rsqrtf(bval / (float)QLR + 1e-6f);
  for (int i = tid; i < QLR; i += 256) out[(size_t)b * QLR + i] = x[i] * rs * w[i];
}

// ---------------------------------------------------------------------------
// Build absorbed q: qcat[b,h, 0:512] = scale * (q_nope[h] . Wk[h])  (bf16)
//                   qcat[b,h,512:576] = scale * rope(q_pe[h], pos=kvlen-1)
// scale = DQ^-0.5 folded in so scores need no post-scale.
// ---------------------------------------------------------------------------
__global__ __launch_bounds__(128) void build_qcat(
    const float* __restrict__ q, const float* __restrict__ kv_b_w,
    bf16_t* __restrict__ qcat) {
  const int h = blockIdx.x, b = blockIdx.y, tid = threadIdx.x;
  __shared__ float qn[DN]; __shared__ float qp[DR];
  const float* qrow = q + ((size_t)b * NH + h) * DQ;
  qn[tid] = qrow[tid];
  if (tid < DR) qp[tid] = qrow[DN + tid];
  __syncthreads();
  const float scale = rsqrtf((float)DQ);
  bf16_t* orow = qcat + ((size_t)b * NH + h) * KCAT;
  for (int c = tid; c < KVLR; c += 128) {
    float acc = 0.f;
    const float* wb = kv_b_w + ((size_t)h * 256) * KVLR + c;
#pragma unroll 4
    for (int d = 0; d < DN; ++d) acc += qn[d] * wb[(size_t)d * KVLR];
    orow[c] = f2bf(acc * scale);
  }
  if (tid < DR / 2) {
    const int i = tid;
    const float freq = __powf(10000.f, -(2.f * i) / (float)DR);
    const float ang  = (float)(KVLEN - 1) * freq;
    const float cs = __cosf(ang), sn = __sinf(ang);
    const float x0 = qp[2 * i], x1 = qp[2 * i + 1];
    orow[KVLR + 2 * i]     = f2bf(scale * (x0 * cs - x1 * sn));
    orow[KVLR + 2 * i + 1] = f2bf(scale * (x0 * sn + x1 * cs));
  }
}

// ---------------------------------------------------------------------------
// Build latent k cache in bf16: kcat[b,k,0:512] = compressed; [512:576] = rope(k_pe).
// One wave per kv row; 128-bit stores.
// ---------------------------------------------------------------------------
__global__ __launch_bounds__(256) void build_kcat(
    const float* __restrict__ comp, const float* __restrict__ kpe,
    bf16_t* __restrict__ kcat) {
  const int row  = blockIdx.x * 8 + (threadIdx.x >> 5); // 0..b*kvlen-1
  const int lane = threadIdx.x & 31;
  const int pos  = row & (KVLEN - 1);
  const float* crow = comp + (size_t)row * KVLR;
  const float* prow = kpe + (size_t)row * DR;
  bf16_t* orow = kcat + (size_t)row * KCAT;
  for (int c = lane * 8; c < KVLR; c += 32 * 8) {
    float4 f0 = *(const float4*)(crow + c);
    float4 f1 = *(const float4*)(crow + c + 4);
    v8bf o;
    o[0] = f2bf(f0.x); o[1] = f2bf(f0.y); o[2] = f2bf(f0.z); o[3] = f2bf(f0.w);
    o[4] = f2bf(f1.x); o[5] = f2bf(f1.y); o[6] = f2bf(f1.z); o[7] = f2bf(f1.w);
    *(v8bf*)(orow + c) = o;
  }
  const int i = lane; // 32 rope pairs, 32 lanes
  const float freq = __powf(10000.f, -(2.f * i) / (float)DR);
  const float ang  = (float)pos * freq;
  const float cs = __cosf(ang), sn = __sinf(ang);
  const float x0 = prow[2 * i], x1 = prow[2 * i + 1];
  orow[KVLR + 2 * i]     = f2bf(x0 * cs - x1 * sn);
  orow[KVLR + 2 * i + 1] = f2bf(x0 * sn + x1 * cs);
}

// ---------------------------------------------------------------------------
// LDS-tiled transpose: kcatT[b,c,kv] = kcat[b,kv,c] for c < 512 (kv-major copy
// of the compressed latents, so the ctx GEMM's B fragments are K-contiguous).
// 64x64 tiles; coalesced reads and writes.
// ---------------------------------------------------------------------------
__global__ __launch_bounds__(256) void kcat_transpose(
    const bf16_t* __restrict__ kcat, bf16_t* __restrict__ kcatT) {
  const int bz = blockIdx.z;
  const int cbase = blockIdx.y * 64;
  const int kvbase = blockIdx.x * 64;
  const int tid = threadIdx.x;
  __shared__ bf16_t tile[64][66]; // 132B row stride: conflict-free
  for (int idx = tid; idx < 64 * 64; idx += 256) {
    const int kvi = idx >> 6, ci = idx & 63;
    tile[kvi][ci] = kcat[((size_t)bz * KVLEN + kvbase + kvi) * KCAT + cbase + ci];
  }
  __syncthreads();
  for (int idx = tid; idx < 64 * 64; idx += 256) {
    const int ci = idx >> 6, kvi = idx & 63;
    kcatT[((size_t)bz * KVLR + cbase + ci) * KVLEN + kvbase + kvi] = tile[kvi][ci];
  }
}

// ---------------------------------------------------------------------------
// scores[b,h,k] = qcat[b,h,:] . kcat[b,k,:]  (M=128 heads, N=2048 kv, K=576)
// WMMA bf16. Block = 4 waves; each wave owns one 16x16 tile; block covers
// 16 heads x 64 kv. A tile (16x576 bf16, 18KB) staged in LDS with b128 copies.
// ---------------------------------------------------------------------------
__global__ __launch_bounds__(128) void scores_wmma(
    const bf16_t* __restrict__ qcat, const bf16_t* __restrict__ kcat,
    float* __restrict__ scores) {
  const int bz = blockIdx.z;
  const int mbase = blockIdx.y * 16;
  const int kvblock = blockIdx.x * 64;
  const int tid = threadIdx.x, wave = tid >> 5, lane = tid & 31;
  const int n = lane & 15, hi = lane >> 4;

  __shared__ __align__(16) bf16_t As[16 * KCAT];
  const bf16_t* qb = qcat + ((size_t)bz * NH + mbase) * KCAT; // 16 contiguous rows
  for (int idx = tid; idx < (16 * KCAT) / 8; idx += 128)      // 128-bit copies
    ((v8bf*)As)[idx] = ((const v8bf*)qb)[idx];
  __syncthreads();

  const int kvrow = kvblock + wave * 16 + n;
  const bf16_t* krow = kcat + ((size_t)bz * KVLEN + kvrow) * KCAT;

  v8f acc = {};
#pragma unroll
  for (int k0 = 0; k0 < KCAT; k0 += 32) {
    // A fragment: row M=lane&15; K chunks {hi*8..+8} and {16+hi*8..+8}
    const bf16_t* ap = &As[(size_t)n * KCAT + k0 + hi * 8];
    v8bf a0 = *reinterpret_cast<const v8bf*>(ap);
    v8bf a1 = *reinterpret_cast<const v8bf*>(ap + 16);
    v16bf A = __builtin_shufflevector(a0, a1, 0,1,2,3,4,5,6,7,8,9,10,11,12,13,14,15);
    // B fragment: col N=lane&15; K chunks {hi*8..+8} and {16+hi*8..+8}
    const bf16_t* bp = krow + k0 + hi * 8;
    v8bf b0 = *reinterpret_cast<const v8bf*>(bp);
    v8bf b1 = *reinterpret_cast<const v8bf*>(bp + 16);
    v16bf B = __builtin_shufflevector(b0, b1, 0,1,2,3,4,5,6,7,8,9,10,11,12,13,14,15);
    acc = __builtin_amdgcn_wmma_f32_16x16x32_bf16(false, A, false, B, (short)0, acc,
                                                  false, false);
  }
  float* srow = scores + ((size_t)bz * NH + mbase) * KVLEN + kvblock + wave * 16 + n;
#pragma unroll
  for (int r = 0; r < 8; ++r) srow[(size_t)(r + hi * 8) * KVLEN] = acc[r];
}

// ---------------------------------------------------------------------------
// Row softmax over kv_len, fp32 in -> bf16 out (scale already folded into q).
// Each thread owns a contiguous 8-element chunk -> b128 loads and stores.
// ---------------------------------------------------------------------------
__global__ __launch_bounds__(256) void softmax_k(
    const float* __restrict__ scores, bf16_t* __restrict__ attn) {
  const int row = blockIdx.x, tid = threadIdx.x;
  const int lane = tid & 31, wid = tid >> 5;
  const float* s = scores + (size_t)row * KVLEN + tid * 8;
  float loc[8]; float mx = -3.4e38f;
  {
    float4 f0 = *(const float4*)(s);
    float4 f1 = *(const float4*)(s + 4);
    loc[0] = f0.x; loc[1] = f0.y; loc[2] = f0.z; loc[3] = f0.w;
    loc[4] = f1.x; loc[5] = f1.y; loc[6] = f1.z; loc[7] = f1.w;
  }
#pragma unroll
  for (int j = 0; j < 8; ++j) mx = fmaxf(mx, loc[j]);
  __shared__ float red[8]; __shared__ float bval;
#pragma unroll
  for (int m = 16; m > 0; m >>= 1) mx = fmaxf(mx, __shfl_xor(mx, m, 32));
  if (lane == 0) red[wid] = mx;
  __syncthreads();
  if (tid == 0) { float t = red[0]; for (int j = 1; j < 8; ++j) t = fmaxf(t, red[j]); bval = t; }
  __syncthreads();
  const float bm = bval;
  float se = 0.f;
#pragma unroll
  for (int j = 0; j < 8; ++j) { loc[j] = __expf(loc[j] - bm); se += loc[j]; }
  se = wave_sum(se);
  __syncthreads();
  if (lane == 0) red[wid] = se;
  __syncthreads();
  if (tid == 0) { float t = 0.f; for (int j = 0; j < 8; ++j) t += red[j]; bval = t; }
  __syncthreads();
  const float inv = 1.f / bval;
  v8bf o;
#pragma unroll
  for (int j = 0; j < 8; ++j) o[j] = f2bf(loc[j] * inv);
  *(v8bf*)(attn + (size_t)row * KVLEN + tid * 8) = o;
}

// ---------------------------------------------------------------------------
// ctx[b,h,c] = attn[b,h,:] . kcatT[b,c,:]  (M=128 heads, N=512 c, K=2048 kv)
// WMMA bf16 with B read from the kv-major transposed cache: both A and B
// fragments are 16B-contiguous global loads. No LDS in the inner loop.
// ---------------------------------------------------------------------------
__global__ __launch_bounds__(128) void ctx_wmma(
    const bf16_t* __restrict__ attn, const bf16_t* __restrict__ kcatT,
    float* __restrict__ ctx) {
  const int bz = blockIdx.z;
  const int mbase = blockIdx.y * 16;
  const int cblock = blockIdx.x * 64;
  const int tid = threadIdx.x, wave = tid >> 5, lane = tid & 31;
  const int n = lane & 15, hi = lane >> 4;

  const int ccol = cblock + wave * 16 + n;
  const bf16_t* arow = attn + ((size_t)bz * NH + mbase + n) * KVLEN;
  const bf16_t* brow = kcatT + ((size_t)bz * KVLR + ccol) * KVLEN;

  v8f acc = {};
#pragma unroll 8
  for (int k0 = 0; k0 < KVLEN; k0 += 32) {
    const bf16_t* ap = arow + k0 + hi * 8;
    v8bf a0 = *reinterpret_cast<const v8bf*>(ap);
    v8bf a1 = *reinterpret_cast<const v8bf*>(ap + 16);
    v16bf A = __builtin_shufflevector(a0, a1, 0,1,2,3,4,5,6,7,8,9,10,11,12,13,14,15);
    const bf16_t* bp = brow + k0 + hi * 8;
    v8bf b0 = *reinterpret_cast<const v8bf*>(bp);
    v8bf b1 = *reinterpret_cast<const v8bf*>(bp + 16);
    v16bf B = __builtin_shufflevector(b0, b1, 0,1,2,3,4,5,6,7,8,9,10,11,12,13,14,15);
    acc = __builtin_amdgcn_wmma_f32_16x16x32_bf16(false, A, false, B, (short)0, acc,
                                                  false, false);
  }
  float* crow = ctx + ((size_t)bz * NH + mbase) * KVLR + ccol;
#pragma unroll
  for (int r = 0; r < 8; ++r) crow[(size_t)(r + hi * 8) * KVLR] = acc[r];
}

// ---------------------------------------------------------------------------
// out[b,h,d] = ctx[b,h,:] . Wv[h,d,:]   (Wv row = kv_b_w[h*256+128+d, :])
// ---------------------------------------------------------------------------
__global__ __launch_bounds__(256) void absorb_v(
    const float* __restrict__ ctx, const float* __restrict__ kv_b_w,
    float* __restrict__ outb) {
  const int h = blockIdx.x, b = blockIdx.y, tid = threadIdx.x;
  const int lane = tid & 31, wid = tid >> 5;
  __shared__ float cx[KVLR];
  const float* crow = ctx + ((size_t)b * NH + h) * KVLR;
  for (int i = tid; i < KVLR; i += 256) cx[i] = crow[i];
  __syncthreads();
  for (int d = wid; d < DV; d += 8) {
    const float* wrow = kv_b_w + ((size_t)(h * 256 + DN + d)) * KVLR;
    float s = 0.f;
    for (int c = lane * 4; c < KVLR; c += 128) {
      float4 wv = *(const float4*)(wrow + c);
      s += wv.x * cx[c] + wv.y * cx[c + 1] + wv.z * cx[c + 2] + wv.w * cx[c + 3];
    }
    s = wave_sum(s);
    if (lane == 0) outb[(size_t)b * (NH * DV) + h * DV + d] = s;
  }
}

// ---------------------------------------------------------------------------

extern "C" void kernel_launch(void* const* d_in, const int* in_sizes, int n_in,
                              void* d_out, int out_size, void* d_ws, size_t ws_size,
                              hipStream_t stream) {
  const float* hidden   = (const float*)d_in[0];
  const float* comp     = (const float*)d_in[1];
  const float* kpe      = (const float*)d_in[2];
  const float* q_a_w    = (const float*)d_in[3];
  const float* q_a_ln_w = (const float*)d_in[4];
  const float* q_b_w    = (const float*)d_in[5];
  const float* kv_b_w   = (const float*)d_in[6];
  const float* o_w      = (const float*)d_in[7];
  float* outp = (float*)d_out;

  char* ws = (char*)d_ws;
  float*  q_a    = (float*) (ws + 0);         // 2*1536 f32
  float*  q_a_n  = (float*) (ws + 12288);     // 2*1536 f32
  float*  q      = (float*) (ws + 24576);     // 2*24576 f32
  bf16_t* qcat   = (bf16_t*)(ws + 221184);    // 2*128*576 bf16
  bf16_t* kcat   = (bf16_t*)(ws + 516096);    // 2*2048*576 bf16
  float*  scores = (float*) (ws + 5234688);   // 2*128*2048 f32
  bf16_t* attn   = (bf16_t*)(ws + 7331840);   // 2*128*2048 bf16
  float*  ctx    = (float*) (ws + 8380416);   // 2*128*512 f32
  float*  outb   = (float*) (ws + 8904704);   // 2*16384 f32
  bf16_t* kcatT  = (bf16_t*)(ws + 9035776);   // 2*512*2048 bf16 (kv-major)

  // 1. q_a = hidden @ q_a_w^T          (2*1536 rows, K=5120)
  gemv_rows<<<(BATCH * QLR) / 8, 256, 0, stream>>>(hidden, q_a_w, q_a, QLR, HDIM);
  // 2. rmsnorm
  rmsnorm_k<<<BATCH, 256, 0, stream>>>(q_a, q_a_ln_w, q_a_n);
  // 3. q = q_a_n @ q_b_w^T             (2*24576 rows, K=1536)
  gemv_rows<<<(BATCH * NH * DQ) / 8, 256, 0, stream>>>(q_a_n, q_b_w, q, NH * DQ, QLR);
  // 4. absorbed q (q_nope.Wk | rope(q_pe)) * DQ^-0.5, bf16
  build_qcat<<<dim3(NH, BATCH), 128, 0, stream>>>(q, kv_b_w, qcat);
  // 5. latent k cache bf16 (compressed | rope(k_pe))
  build_kcat<<<(BATCH * KVLEN) / 8, 256, 0, stream>>>(comp, kpe, kcat);
  // 5b. kv-major transposed copy of the compressed latents for the ctx GEMM
  kcat_transpose<<<dim3(KVLEN / 64, KVLR / 64, BATCH), 256, 0, stream>>>(kcat, kcatT);
  // 6. scores = qcat @ kcat^T  (WMMA bf16, K=576)
  scores_wmma<<<dim3(KVLEN / 64, NH / 16, BATCH), 128, 0, stream>>>(qcat, kcat, scores);
  // 7. softmax -> attn (bf16)
  softmax_k<<<BATCH * NH, 256, 0, stream>>>(scores, attn);
  // 8. ctx = attn @ compressed  (WMMA bf16, K=2048, B from kcatT)
  ctx_wmma<<<dim3(KVLR / 64, NH / 16, BATCH), 128, 0, stream>>>(attn, kcatT, ctx);
  // 9. out = ctx absorbed with Wv
  absorb_v<<<dim3(NH, BATCH), 256, 0, stream>>>(ctx, kv_b_w, outb);
  // 10. final = out @ o_w^T            (2*5120 rows, K=16384) — biggest stream, 335MB
  gemv_rows<<<(BATCH * HDIM) / 8, 256, 0, stream>>>(outb, o_w, outp, HDIM, NH * DV);
}
